// VAE_17961553232296
// MI455X (gfx1250) — matrix operand order repaired and checked by reading, not compiled, not run on hardware
//
#include <hip/hip_runtime.h>
#include <cstdint>
#include <cstddef>

// ---------------- problem constants ----------------
static constexpr int Bq    = 4;
static constexpr int Nn    = 256;
static constexpr int BN    = Bq * Nn;        // 1024 node rows
static constexpr int BNN_  = Bq * Nn * Nn;   // 262144 edge rows
static constexpr int HEADS = 8;

// ---------------- WMMA types (CDNA5, wave32) ----------------
typedef __attribute__((ext_vector_type(16))) __bf16 v16bf;
typedef __attribute__((ext_vector_type(8)))  float  v8f;

union AFrag { v16bf v; unsigned u[8]; };

// Native fp32->bf16 (RNE) pack: lowers to v_cvt_pk_bf16_f32 on gfx1250.
__device__ __forceinline__ unsigned pk2(float lo, float hi) {
  union { __bf16 h[2]; unsigned u; } t;
  t.h[0] = (__bf16)lo;
  t.h[1] = (__bf16)hi;
  return t.u;
}
__device__ __forceinline__ v8f zero8() {
  v8f z;
#pragma unroll
  for (int i = 0; i < 8; ++i) z[i] = 0.f;
  return z;
}

// =====================================================================
// Weight pre-pack: W (K,N) fp32 -> P (K/2, N) u32 of packed bf16 pairs
// P[p*N + n] = pack(W[2p][n], W[2p+1][n])  == WMMA B-frag natural layout.
// =====================================================================
__global__ __launch_bounds__(256) void pack_kernel(
    const float* __restrict__ W, unsigned* __restrict__ P, int KP, int N) {
  int id = blockIdx.x * 256 + threadIdx.x;
  if (id >= KP * N) return;
  int p = id / N, n = id - p * N;
  P[(size_t)p * N + n] =
      pk2(W[(size_t)(2 * p) * N + n], W[(size_t)(2 * p + 1) * N + n]);
}

// =====================================================================
// Generic GEMM: Y[M,NOUT] = X[M,K] @ W[K,NOUT] (+bias) (+=Y residual)
// K and NOUT are compile-time: B-frag loads become base + immediate
// offsets, LDS strides fold to constants. X fp32 -> bf16 while staging
// the 16-row A tile into LDS (A staged fully before any global write,
// so in-place row-preserving GEMMs (Y==X) are safe). W pre-packed bf16.
// One WG = 4 waves = one 16-row M tile; wave covers 64-wide N strips.
// A-frag (16-bit A 16x32): lane l: m=l&15, lh=l>>4;
//   u[r]=pair(lh*8+r)
// B-frag (16-bit B 32x16): lane l: n=l&15, g=l>>4; u[r]: pair g*8+r
// D (f32 16x16): lane l: n=l&15, lh=l>>4; acc[r] -> row lh*8+r.
// =====================================================================
template <int K, int NOUT, bool RESID, bool HASBIAS>
__global__ __launch_bounds__(128) void gemm_kernel(
    const float* __restrict__ X, const unsigned* __restrict__ Wp,
    const float* __restrict__ bias, float* __restrict__ Y) {
  constexpr int KP = K / 2;                 // bf16 pairs per row
  __shared__ unsigned lds_a[16 * KP];
  const int m0  = blockIdx.x * 16;
  const int tid = threadIdx.x;

  for (int idx = tid; idx < 16 * KP; idx += 128) {
    int r = idx / KP, p = idx - r * KP;
    const float* src = X + (size_t)(m0 + r) * K + 2 * p;
    lds_a[r * KP + p] = pk2(src[0], src[1]);
  }
  __syncthreads();

  const int wave = tid >> 5;
  const int lane = tid & 31;
  const int lm   = lane & 15;
  const int lh   = lane >> 4;

  for (int n0 = wave * 64; n0 < NOUT; n0 += 256) {
    v8f acc[4];
#pragma unroll
    for (int c = 0; c < 4; ++c) acc[c] = zero8();

    for (int k0 = 0; k0 < K; k0 += 32) {
      AFrag a;
      const unsigned* arow = lds_a + lm * KP + (k0 >> 1);
      uint4 alo = *reinterpret_cast<const uint4*>(arow + lh * 8);
      uint4 ahi = *reinterpret_cast<const uint4*>(arow + lh * 8 + 4);
      a.u[0] = alo.x; a.u[1] = alo.y; a.u[2] = alo.z; a.u[3] = alo.w;
      a.u[4] = ahi.x; a.u[5] = ahi.y; a.u[6] = ahi.z; a.u[7] = ahi.w;

#pragma unroll
      for (int c = 0; c < 4; ++c) {
        int nc = n0 + c * 16 + lm;
        AFrag b;
        if constexpr (NOUT % 64 == 0) {
          const unsigned* wp = Wp + (size_t)((k0 >> 1) + lh * 8) * NOUT + nc;
#pragma unroll
          for (int r = 0; r < 8; ++r) b.u[r] = wp[r * NOUT];
        } else {
          if (nc < NOUT) {
            const unsigned* wp = Wp + (size_t)((k0 >> 1) + lh * 8) * NOUT + nc;
#pragma unroll
            for (int r = 0; r < 8; ++r) b.u[r] = wp[r * NOUT];
          } else {
#pragma unroll
            for (int r = 0; r < 8; ++r) b.u[r] = 0u;
          }
        }
        acc[c] = __builtin_amdgcn_wmma_f32_16x16x32_bf16(
            false, a.v, false, b.v, (short)0, acc[c], false, false);
      }
    }
#pragma unroll
    for (int c = 0; c < 4; ++c) {
      int col = n0 + c * 16 + lm;
      if constexpr (NOUT % 64 != 0) { if (col >= NOUT) continue; }
      float bv = HASBIAS ? bias[col] : 0.f;
#pragma unroll
      for (int r = 0; r < 8; ++r) {
        int row = m0 + lh * 8 + r;          // M is always a multiple of 16
        size_t off = (size_t)row * NOUT + col;
        float v = acc[c][r] + bv;
        if (RESID) v += Y[off];
        Y[off] = v;
      }
    }
  }
}

template <int K, int NOUT>
static void launch_gemm(hipStream_t s, const float* X, const unsigned* Wp,
                        const float* bias, float* Y, int M, bool resid) {
  dim3 g(M / 16), b(128);
  if (resid)
    gemm_kernel<K, NOUT, true, true><<<g, b, 0, s>>>(X, Wp, bias, Y);
  else if (bias)
    gemm_kernel<K, NOUT, false, true><<<g, b, 0, s>>>(X, Wp, bias, Y);
  else
    gemm_kernel<K, NOUT, false, false><<<g, b, 0, s>>>(X, Wp, bias, Y);
}

// =====================================================================
// LayerNorm over 256 features, one row per 256-thread block. RELU opt.
// =====================================================================
template <bool RELU>
__global__ __launch_bounds__(256) void ln256_kernel(
    const float* __restrict__ X, float* __restrict__ Y,
    const float* __restrict__ g, const float* __restrict__ b) {
  __shared__ float sm[8];
  __shared__ float bc;
  size_t row = blockIdx.x;
  int d = threadIdx.x;
  float x = X[row * 256 + d];
  float s = x;
  for (int o = 16; o > 0; o >>= 1) s += __shfl_xor(s, o, 32);
  if ((d & 31) == 0) sm[d >> 5] = s;
  __syncthreads();
  if (d == 0) { float t = 0; for (int i = 0; i < 8; ++i) t += sm[i]; bc = t * (1.f / 256.f); }
  __syncthreads();
  float mean = bc;
  float c = x - mean;
  float s2 = c * c;
  for (int o = 16; o > 0; o >>= 1) s2 += __shfl_xor(s2, o, 32);
  __syncthreads();
  if ((d & 31) == 0) sm[d >> 5] = s2;
  __syncthreads();
  if (d == 0) { float t = 0; for (int i = 0; i < 8; ++i) t += sm[i]; bc = rsqrtf(t * (1.f / 256.f) + 1e-5f); }
  __syncthreads();
  float y = c * bc * g[d] + b[d];
  if (RELU) y = fmaxf(y, 0.f);
  Y[row * 256 + d] = y;
}

// Edge-bias LN over 8 heads, in place.
__global__ __launch_bounds__(256) void ln8_kernel(
    float* __restrict__ EB, const float* __restrict__ g,
    const float* __restrict__ b) {
  int r = blockIdx.x * 256 + threadIdx.x;
  if (r >= BNN_) return;
  float v[8]; float m = 0.f;
#pragma unroll
  for (int i = 0; i < 8; ++i) { v[i] = EB[(size_t)r * 8 + i]; m += v[i]; }
  m *= 0.125f;
  float var = 0.f;
#pragma unroll
  for (int i = 0; i < 8; ++i) { float c = v[i] - m; var += c * c; }
  float rs = rsqrtf(var * 0.125f + 1e-5f);
#pragma unroll
  for (int i = 0; i < 8; ++i) EB[(size_t)r * 8 + i] = (v[i] - m) * rs * g[i] + b[i];
}

// Build pad mask and encoder (pad+knn) mask, (B,N,N) each.
__global__ __launch_bounds__(256) void mask_kernel(
    const unsigned char* __restrict__ mask, const unsigned char* __restrict__ knn,
    float* __restrict__ ENCM, float* __restrict__ PADM) {
  size_t r = (size_t)blockIdx.x * 256 + threadIdx.x;
  if (r >= (size_t)BNN_) return;
  int bidx = (int)(r >> 16);
  int i = (int)((r >> 8) & 255);
  int j = (int)(r & 255);
  float mi = mask[bidx * 256 + i] ? 0.f : -1e9f;
  float mj = mask[bidx * 256 + j] ? 0.f : -1e9f;
  float pad = mi + mj;
  PADM[r] = pad;
  ENCM[r] = pad + (knn[r] ? 0.f : -1e9f);
}

// =====================================================================
// scores[b,h,i,j] = (q_i.k_j)/sqrt(32) + eb[b,i,j,h] + AM[b,i,j]
// One (b,h) per blockIdx.y; 16 i-rows per block, wave w covers j-strip.
// =====================================================================
__global__ __launch_bounds__(128) void scores_kernel(
    const float* __restrict__ QKV, const float* __restrict__ EB,
    const float* __restrict__ AM, float* __restrict__ S) {
  int bh = blockIdx.y;
  int b = bh >> 3, h = bh & 7;
  int i0 = blockIdx.x * 16;
  int wave = threadIdx.x >> 5, lane = threadIdx.x & 31;
  int lm = lane & 15, lh = lane >> 4;
  int n0 = wave * 64;

  AFrag a;
  const float* qrow = QKV + ((size_t)(b * 256 + i0 + lm)) * 768 + h * 32;
#pragma unroll
  for (int r = 0; r < 4; ++r) a.u[r]     = pk2(qrow[lh * 8 + 2 * r], qrow[lh * 8 + 2 * r + 1]);
#pragma unroll
  for (int r = 0; r < 4; ++r) a.u[4 + r] = pk2(qrow[16 + lh * 8 + 2 * r], qrow[16 + lh * 8 + 2 * r + 1]);

  v8f acc[4];
#pragma unroll
  for (int c = 0; c < 4; ++c) acc[c] = zero8();

#pragma unroll
  for (int c = 0; c < 4; ++c) {
    int j = n0 + c * 16 + lm;
    const float* krow = QKV + ((size_t)(b * 256 + j)) * 768 + 256 + h * 32;
    AFrag bb;
#pragma unroll
    for (int r = 0; r < 8; ++r)
      bb.u[r] = pk2(krow[lh * 8 + 2 * r + (r >= 4 ? 8 : 0)], krow[lh * 8 + 2 * r + (r >= 4 ? 8 : 0) + 1]);
    acc[c] = __builtin_amdgcn_wmma_f32_16x16x32_bf16(
        false, a.v, false, bb.v, (short)0, acc[c], false, false);
  }

#pragma unroll
  for (int c = 0; c < 4; ++c) {
    int j = n0 + c * 16 + lm;
#pragma unroll
    for (int r = 0; r < 8; ++r) {
      int i = i0 + lh * 8 + r;
      float s = acc[c][r] * 0.17677669529663687f
              + EB[(((size_t)(b * 256 + i)) * 256 + j) * 8 + h]
              + AM[((size_t)b * 256 + i) * 256 + j];
      S[(((size_t)bh) * 256 + i) * 256 + j] = s;
    }
  }
}

__global__ __launch_bounds__(256) void softmax_kernel(float* __restrict__ S) {
  __shared__ float sm[8];
  __shared__ float bc;
  size_t row = blockIdx.x;
  int j = threadIdx.x;
  float v = S[row * 256 + j];
  float m = v;
  for (int o = 16; o > 0; o >>= 1) m = fmaxf(m, __shfl_xor(m, o, 32));
  if ((j & 31) == 0) sm[j >> 5] = m;
  __syncthreads();
  if (j == 0) { float t = sm[0]; for (int i = 1; i < 8; ++i) t = fmaxf(t, sm[i]); bc = t; }
  __syncthreads();
  float e = __expf(v - bc);
  float s = e;
  for (int o = 16; o > 0; o >>= 1) s += __shfl_xor(s, o, 32);
  __syncthreads();
  if ((j & 31) == 0) sm[j >> 5] = s;
  __syncthreads();
  if (j == 0) { float t = 0; for (int i = 0; i < 8; ++i) t += sm[i]; bc = t; }
  __syncthreads();
  S[row * 256 + j] = e / bc;
}

// h_out[b,i,h,:] = probs[b,h,i,:] @ V[b,:,h,:]   (K=256, Nout=32)
__global__ __launch_bounds__(32) void pv_kernel(
    const float* __restrict__ S, const float* __restrict__ QKV,
    float* __restrict__ HO) {
  int bh = blockIdx.y;
  int b = bh >> 3, h = bh & 7;
  int i0 = blockIdx.x * 16;
  int lane = threadIdx.x;
  int lm = lane & 15, lh = lane >> 4;

  v8f acc[2];
  acc[0] = zero8(); acc[1] = zero8();

  for (int k0 = 0; k0 < 256; k0 += 32) {
    AFrag a;
    const float* prow = S + (((size_t)bh) * 256 + i0 + lm) * 256 + k0;
#pragma unroll
    for (int r = 0; r < 4; ++r) a.u[r]     = pk2(prow[lh * 8 + 2 * r], prow[lh * 8 + 2 * r + 1]);
#pragma unroll
    for (int r = 0; r < 4; ++r) a.u[4 + r] = pk2(prow[16 + lh * 8 + 2 * r], prow[16 + lh * 8 + 2 * r + 1]);

#pragma unroll
    for (int c = 0; c < 2; ++c) {
      int n = c * 16 + lm;
      AFrag bb;
#pragma unroll
      for (int r = 0; r < 8; ++r) {
        int kk = k0 + lh * 8 + 2 * r + (r >= 4 ? 8 : 0);
        const float* vp = QKV + ((size_t)(b * 256 + kk)) * 768 + 512 + h * 32 + n;
        bb.u[r] = pk2(vp[0], vp[768]);
      }
      acc[c] = __builtin_amdgcn_wmma_f32_16x16x32_bf16(
          false, a.v, false, bb.v, (short)0, acc[c], false, false);
    }
  }
#pragma unroll
  for (int c = 0; c < 2; ++c) {
    int col = c * 16 + lm;
#pragma unroll
    for (int r = 0; r < 8; ++r) {
      int row = i0 + lh * 8 + r;
      HO[((size_t)(b * 256 + row)) * 256 + h * 32 + col] = acc[c][r];
    }
  }
}

// hid = LN(relu(hi[b,i]+hj[b,j]+ee)), in place on E (one edge row / block)
__global__ __launch_bounds__(256) void edge_mlp_kernel(
    const float* __restrict__ HI, const float* __restrict__ HJ,
    float* __restrict__ E, const float* __restrict__ g,
    const float* __restrict__ b) {
  __shared__ float sm[8];
  __shared__ float bc;
  size_t r = blockIdx.x;
  int d = threadIdx.x;
  size_t bidx = r >> 16;
  int i = (int)((r >> 8) & 255);
  int j = (int)(r & 255);
  float x = E[r * 256 + d]
          + HI[(bidx * 256 + i) * 256 + d]
          + HJ[(bidx * 256 + j) * 256 + d];
  x = fmaxf(x, 0.f);
  float s = x;
  for (int o = 16; o > 0; o >>= 1) s += __shfl_xor(s, o, 32);
  if ((d & 31) == 0) sm[d >> 5] = s;
  __syncthreads();
  if (d == 0) { float t = 0; for (int k = 0; k < 8; ++k) t += sm[k]; bc = t * (1.f / 256.f); }
  __syncthreads();
  float mean = bc;
  float c = x - mean;
  float s2 = c * c;
  for (int o = 16; o > 0; o >>= 1) s2 += __shfl_xor(s2, o, 32);
  __syncthreads();
  if ((d & 31) == 0) sm[d >> 5] = s2;
  __syncthreads();
  if (d == 0) { float t = 0; for (int k = 0; k < 8; ++k) t += sm[k]; bc = rsqrtf(t * (1.f / 256.f) + 1e-5f); }
  __syncthreads();
  E[r * 256 + d] = c * bc * g[d] + b[d];
}

// e_dec[b,i,j,:] = za[b,i,:] + zb[b,j,:] + bias
__global__ __launch_bounds__(256) void zedge_kernel(
    const float* __restrict__ ZA, const float* __restrict__ ZB,
    const float* __restrict__ bias, float* __restrict__ E) {
  size_t r = blockIdx.x;
  int d = threadIdx.x;
  size_t bidx = r >> 16;
  int i = (int)((r >> 8) & 255);
  int j = (int)(r & 255);
  E[r * 256 + d] = ZA[(bidx * 256 + i) * 256 + d]
                 + ZB[(bidx * 256 + j) * 256 + d] + bias[d];
}

// logvar clip (in place) + z = mu + eps*exp(0.5*logvar)
__global__ __launch_bounds__(256) void z_kernel(
    const float* __restrict__ mu, float* __restrict__ logvar,
    const float* __restrict__ eps, float* __restrict__ z, int n) {
  int i = blockIdx.x * 256 + threadIdx.x;
  if (i >= n) return;
  float lv = fminf(fmaxf(logvar[i], -10.f), 10.f);
  logvar[i] = lv;
  z[i] = mu[i] + eps[i] * __expf(0.5f * lv);
}

// rot6d -> 3x3 rotation (Gram-Schmidt)
__global__ __launch_bounds__(256) void rot_kernel(
    const float* __restrict__ R6, float* __restrict__ R) {
  int r = blockIdx.x * 256 + threadIdx.x;
  if (r >= BN) return;
  float a1x = R6[r * 6 + 0], a1y = R6[r * 6 + 1], a1z = R6[r * 6 + 2];
  float a2x = R6[r * 6 + 3], a2y = R6[r * 6 + 4], a2z = R6[r * 6 + 5];
  float n1 = fmaxf(sqrtf(a1x * a1x + a1y * a1y + a1z * a1z), 1e-8f);
  float b1x = a1x / n1, b1y = a1y / n1, b1z = a1z / n1;
  float dt = a2x * b1x + a2y * b1y + a2z * b1z;
  float cx = a2x - dt * b1x, cy = a2y - dt * b1y, cz = a2z - dt * b1z;
  float n2 = fmaxf(sqrtf(cx * cx + cy * cy + cz * cz), 1e-8f);
  float b2x = cx / n2, b2y = cy / n2, b2z = cz / n2;
  float b3x = b1y * b2z - b1z * b2y;
  float b3y = b1z * b2x - b1x * b2z;
  float b3z = b1x * b2y - b1y * b2x;
  float* o = R + (size_t)r * 9;
  o[0] = b1x; o[1] = b1y; o[2] = b1z;
  o[3] = b2x; o[4] = b2y; o[5] = b2z;
  o[6] = b3x; o[7] = b3y; o[8] = b3z;
}

// =====================================================================
// Host side
// =====================================================================
struct BlkP {
  const float *eb_b, *eb_w, *ebln_b, *ebln_g, *em1_b, *em1_w, *em2_b, *em2_w,
              *emln_b, *emln_g, *ne_b, *ne_g, *nh_b, *nh_g, *out_b, *out_w,
              *qkv_b, *qkv_w;
  // packed bf16 weights (filled at launch)
  const unsigned *qkv_p, *eb_p, *em1_p, *em2_p, *out_p;
};

extern "C" void kernel_launch(void* const* d_in, const int* in_sizes, int n_in,
                              void* d_out, int out_size, void* d_ws, size_t ws_size,
                              hipStream_t stream) {
  (void)in_sizes; (void)n_in; (void)out_size; (void)ws_size;

  // ----- inputs (setup_inputs insertion order; params pytree-flattened) -----
  const float* h_in   = (const float*)d_in[0];   // (B,N,128)
  const float* e_in   = (const float*)d_in[1];   // (B,N,N,64)
  const float* eps_in = (const float*)d_in[2];   // (B,N,64)

  int idx = 3;
  auto nx = [&]() { return (const float*)d_in[idx++]; };
  auto loadBlk = [&](BlkP& p) {
    // blk dict sorted: eb, eb_ln, em1, em2, em_ln, norm_e, norm_h, out, qkv
    // W = {b,w} sorted; LN = {b,g} sorted
    p.eb_b = nx();  p.eb_w = nx();
    p.ebln_b = nx(); p.ebln_g = nx();
    p.em1_b = nx(); p.em1_w = nx();
    p.em2_b = nx(); p.em2_w = nx();
    p.emln_b = nx(); p.emln_g = nx();
    p.ne_b = nx(); p.ne_g = nx();
    p.nh_b = nx(); p.nh_g = nx();
    p.out_b = nx(); p.out_w = nx();
    p.qkv_b = nx(); p.qkv_w = nx();
  };
  // params sorted keys: dec, dec_z_edge, dec_z_in, edge_in, enc, fa1, fa2,
  //                     fa_ln, out_node, out_rot, out_trans, to_logvar, to_mu
  BlkP dec[4]; for (int i = 0; i < 4; ++i) loadBlk(dec[i]);
  const float *dze_b = nx(), *dze_w = nx();
  const float *dzi_b = nx(), *dzi_w = nx();
  const float *ein_b = nx(), *ein_w = nx();
  BlkP enc[4]; for (int i = 0; i < 4; ++i) loadBlk(enc[i]);
  const float *fa1_b = nx(), *fa1_w = nx();
  const float *fa2_b = nx(), *fa2_w = nx();
  const float *faln_b = nx(), *faln_g = nx();
  const float *on_b = nx(), *on_w = nx();
  const float *orot_b = nx(), *orot_w = nx();
  const float *ot_b = nx(), *ot_w = nx();
  const float *tlv_b = nx(), *tlv_w = nx();
  const float *tmu_b = nx(), *tmu_w = nx();
  const unsigned char* maskp = (const unsigned char*)d_in[idx++];
  const unsigned char* knnp  = (const unsigned char*)d_in[idx++];

  // ----- outputs (flat, return order) -----
  float* out  = (float*)d_out;
  float* o_hp = out;                    // h_pred   (BN,256)
  float* o_tp = out + 262144;           // t_pred   (BN,3)
  float* o_rp = out + 265216;           // r_pred   (BN,9)
  float* o_mu = out + 274432;           // mu       (BN,64)
  float* o_lv = out + 339968;           // logvar   (BN,64)
  float* o_z  = out + 405504;           // z        (BN,64)
  float* o_ht = out + 471040;           // h_target (BN,256)

  // ----- workspace -----
  float* Wf = (float*)d_ws;
  size_t off = 0;
  auto alloc = [&](size_t n) { float* p = Wf + off; off += n; return p; };
  float* E0   = alloc((size_t)BNN_ * 256);   // edge state
  float* E1   = alloc((size_t)BNN_ * 256);   // normed e / hid
  float* EBv  = alloc((size_t)BNN_ * 8);     // edge bias per head
  float* Sv   = alloc((size_t)BNN_ * 8);     // scores/probs (B,H,N,N)
  float* ENCM = alloc(BNN_);
  float* PADM = alloc(BNN_);
  float* X0   = alloc((size_t)BN * 256);
  float* X1   = alloc((size_t)BN * 256);
  float* Hb   = alloc((size_t)BN * 256);     // node state
  float* HN   = alloc((size_t)BN * 256);
  float* QKVb = alloc((size_t)BN * 768);
  float* HO   = alloc((size_t)BN * 256);
  float* HI   = alloc((size_t)BN * 256);     // also za
  float* HJ   = alloc((size_t)BN * 256);     // also zb
  float* R6   = alloc((size_t)BN * 6);

  // pre-packed bf16 weights (u32 pairs), same 4-byte granularity as float
  auto pack = [&](const float* w, int K, int N) -> const unsigned* {
    unsigned* dst = (unsigned*)alloc((size_t)(K / 2) * N);
    int tot = (K / 2) * N;
    pack_kernel<<<(tot + 255) / 256, 256, 0, stream>>>(w, dst, K / 2, N);
    return dst;
  };
  auto packBlk = [&](BlkP& p) {
    p.qkv_p = pack(p.qkv_w, 256, 768);
    p.eb_p  = pack(p.eb_w, 256, 8);
    p.em1_p = pack(p.em1_w, 768, 256);   // hi/hj/ee slices at pair offsets
    p.em2_p = pack(p.em2_w, 256, 256);
    p.out_p = pack(p.out_w, 256, 256);
  };
  for (int i = 0; i < 4; ++i) packBlk(enc[i]);
  for (int i = 0; i < 4; ++i) packBlk(dec[i]);
  const unsigned* fa1_p  = pack(fa1_w, 128, 256);
  const unsigned* fa2_p  = pack(fa2_w, 256, 256);
  const unsigned* ein_p  = pack(ein_w, 64, 256);
  const unsigned* tmu_p  = pack(tmu_w, 256, 64);
  const unsigned* tlv_p  = pack(tlv_w, 256, 64);
  const unsigned* dzi_p  = pack(dzi_w, 64, 256);
  const unsigned* dze_p  = pack(dze_w, 128, 256); // za rows 0..63 -> pairs 0..31
  const unsigned* on_p   = pack(on_w, 256, 256);
  const unsigned* orot_p = pack(orot_w, 256, 6);
  const unsigned* ot_p   = pack(ot_w, 256, 3);

  auto attn_layer = [&](const BlkP& p, const float* AM) {
    ln256_kernel<false><<<BN, 256, 0, stream>>>(Hb, HN, p.nh_g, p.nh_b);
    launch_gemm<256, 768>(stream, HN, p.qkv_p, p.qkv_b, QKVb, BN, false);
    ln256_kernel<false><<<BNN_, 256, 0, stream>>>(E0, E1, p.ne_g, p.ne_b);
    launch_gemm<256, 8>(stream, E1, p.eb_p, p.eb_b, EBv, BNN_, false);
    ln8_kernel<<<BNN_ / 256, 256, 0, stream>>>(EBv, p.ebln_g, p.ebln_b);
    scores_kernel<<<dim3(16, Bq * HEADS), 128, 0, stream>>>(QKVb, EBv, AM, Sv);
    softmax_kernel<<<Bq * HEADS * Nn, 256, 0, stream>>>(Sv);
    pv_kernel<<<dim3(16, Bq * HEADS), 32, 0, stream>>>(Sv, QKVb, HO);
    launch_gemm<256, 256>(stream, HO, p.out_p, p.out_b, Hb, BN, true);  // h += out-proj
    launch_gemm<256, 256>(stream, Hb, p.em1_p,             nullptr, HI, BN, false);
    launch_gemm<256, 256>(stream, Hb, p.em1_p + 128 * 256, nullptr, HJ, BN, false);
    launch_gemm<256, 256>(stream, E1, p.em1_p + 256 * 256, p.em1_b, E1, BNN_, false); // ee in place
    edge_mlp_kernel<<<BNN_, 256, 0, stream>>>(HI, HJ, E1, p.emln_g, p.emln_b);
    launch_gemm<256, 256>(stream, E1, p.em2_p, p.em2_b, E0, BNN_, true); // e += em2(hid)
  };

  // ----- masks -----
  mask_kernel<<<BNN_ / 256, 256, 0, stream>>>(maskp, knnp, ENCM, PADM);

  // ----- feature head: h_target -----
  launch_gemm<128, 256>(stream, h_in, fa1_p, fa1_b, X0, BN, false);
  ln256_kernel<true><<<BN, 256, 0, stream>>>(X0, X1, faln_g, faln_b);
  launch_gemm<256, 256>(stream, X1, fa2_p, fa2_b, o_ht, BN, false);
  hipMemcpyAsync(Hb, o_ht, (size_t)BN * 256 * sizeof(float),
                 hipMemcpyDeviceToDevice, stream);

  // ----- edge embed + encoder -----
  launch_gemm<64, 256>(stream, e_in, ein_p, ein_b, E0, BNN_, false);
  for (int L = 0; L < 4; ++L) attn_layer(enc[L], ENCM);

  // ----- latent -----
  launch_gemm<256, 64>(stream, Hb, tmu_p, tmu_b, o_mu, BN, false);
  launch_gemm<256, 64>(stream, Hb, tlv_p, tlv_b, o_lv, BN, false);
  z_kernel<<<(BN * 64) / 256, 256, 0, stream>>>(o_mu, o_lv, eps_in, o_z, BN * 64);

  // ----- decoder init -----
  launch_gemm<64, 256>(stream, o_z, dzi_p, dzi_b, Hb, BN, false);
  launch_gemm<64, 256>(stream, o_z, dze_p,            nullptr, HI, BN, false); // za
  launch_gemm<64, 256>(stream, o_z, dze_p + 32 * 256, nullptr, HJ, BN, false); // zb
  zedge_kernel<<<BNN_, 256, 0, stream>>>(HI, HJ, dze_b, E0);
  for (int L = 0; L < 4; ++L) attn_layer(dec[L], PADM);

  // ----- output heads -----
  launch_gemm<256, 256>(stream, Hb, on_p, on_b, o_hp, BN, false);
  launch_gemm<256, 3>(stream, Hb, ot_p, ot_b, o_tp, BN, false);
  launch_gemm<256, 6>(stream, Hb, orot_p, orot_b, R6, BN, false);
  rot_kernel<<<(BN + 255) / 256, 256, 0, stream>>>(R6, o_rp);
}
